// Model_48644799594552
// MI455X (gfx1250) — matrix operand order repaired and checked by reading, not compiled, not run on hardware
//
#include <hip/hip_runtime.h>
#include <hip/hip_bf16.h>
#include <math.h>

// ---------------- problem constants ----------------
#define B_SZ 4
#define L_SEQ 2048
#define C_IN 7
#define C_OUT 7
#define D_MODEL 512
#define D_INNER 1024
#define D_STATE 16
#define D_CONV 4
#define DT_RANK 32
#define N_MARK 4
#define PRED_LEN 96
#define T_OFF (L_SEQ - PRED_LEN)   // 1952
#define M_ROWS (B_SZ * L_SEQ)      // 8192
#define M_TAIL (B_SZ * PRED_LEN)   // 384

typedef __attribute__((ext_vector_type(8)))  __bf16 v8bf;
typedef __attribute__((ext_vector_type(16))) __bf16 v16bf;
typedef __attribute__((ext_vector_type(8)))  float  v8f;
typedef unsigned short u16t;
typedef unsigned int   u32t;

__device__ __forceinline__ u16t f2bf(float f) {
  u32t u = __float_as_uint(f);
  u += 0x7FFFu + ((u >> 16) & 1u);      // round-to-nearest-even
  return (u16t)(u >> 16);
}
__device__ __forceinline__ float sigmoidf_(float x) { return 1.0f / (1.0f + __expf(-x)); }
__device__ __forceinline__ float siluf_(float x)    { return x * sigmoidf_(x); }
__device__ __forceinline__ float softplusf_(float x){ return (x > 20.f) ? x : log1pf(__expf(x)); }

// A fragment: 16 contiguous K values at p[0..15]
__device__ __forceinline__ v16bf load_a_frag(const u16t* p) {
  v8bf lo = *reinterpret_cast<const v8bf*>(p);
  v8bf hi = *reinterpret_cast<const v8bf*>(p + 8);
  return __builtin_shufflevector(lo, hi, 0,1,2,3,4,5,6,7,8,9,10,11,12,13,14,15);
}
// B fragment: 16 contiguous K values
__device__ __forceinline__ v16bf load_b_frag(const u16t* p) {
  v8bf lo = *reinterpret_cast<const v8bf*>(p);
  v8bf hi = *reinterpret_cast<const v8bf*>(p + 8);
  return __builtin_shufflevector(lo, hi, 0,1,2,3,4,5,6,7,8,9,10,11,12,13,14,15);
}

// ---------------- instance-norm stats (mean/std per (b,c)) ----------------
__global__ void instnorm_stats(const float* __restrict__ x_enc,
                               float* __restrict__ meanv, float* __restrict__ stdv) {
  __shared__ float ss[256], sq[256];
  int bc = blockIdx.x; int b = bc / C_IN, c = bc % C_IN;
  float s = 0.f, q = 0.f;
  for (int t = threadIdx.x; t < L_SEQ; t += 256) {
    float v = x_enc[((size_t)b * L_SEQ + t) * C_IN + c];
    s += v; q += v * v;
  }
  ss[threadIdx.x] = s; sq[threadIdx.x] = q;
  __syncthreads();
  for (int st = 128; st > 0; st >>= 1) {
    if (threadIdx.x < st) { ss[threadIdx.x] += ss[threadIdx.x + st]; sq[threadIdx.x] += sq[threadIdx.x + st]; }
    __syncthreads();
  }
  if (threadIdx.x == 0) {
    float m = ss[0] / (float)L_SEQ;
    float var = sq[0] / (float)L_SEQ - m * m;
    meanv[bc] = m;
    stdv[bc]  = sqrtf(var + 1e-5f);
  }
}

// ---------------- f32 -> bf16 converts ----------------
__global__ void f32_to_bf16(const float* __restrict__ src, u16t* __restrict__ dst, int n) {
  int i = blockIdx.x * blockDim.x + threadIdx.x;
  if (i < n) dst[i] = f2bf(src[i]);
}
// take first 32 cols of a 64-col row-major matrix -> compact bf16
__global__ void slice32of64_to_bf16(const float* __restrict__ src, u16t* __restrict__ dst, int n) {
  int i = blockIdx.x * blockDim.x + threadIdx.x;
  if (i < n) { int m = i >> 5, j = i & 31; dst[i] = f2bf(src[(size_t)m * 64 + j]); }
}

// ---------------- fused embedding: circular conv1d(k=3) + temporal + positional, bf16 out ----------------
__global__ void embed_kernel(const float* __restrict__ x_enc, const float* __restrict__ x_mark,
                             const float* __restrict__ token_w, const float* __restrict__ temp_w,
                             const float* __restrict__ meanv, const float* __restrict__ stdv,
                             u16t* __restrict__ xemb_bf) {
  int bt = blockIdx.x; int b = bt / L_SEQ, t = bt % L_SEQ;
  int d = threadIdx.x;                         // 512 threads, one per model dim
  __shared__ float s_xn[3 * C_IN];
  __shared__ float s_mk[N_MARK];
  if (d < 3 * C_IN) {
    int k = d / C_IN, c = d % C_IN;
    int tt = t + k - 1; tt = (tt < 0) ? tt + L_SEQ : (tt >= L_SEQ ? tt - L_SEQ : tt);
    s_xn[d] = (x_enc[((size_t)b * L_SEQ + tt) * C_IN + c] - meanv[b * C_IN + c]) / stdv[b * C_IN + c];
  }
  if (d >= 32 && d < 32 + N_MARK)
    s_mk[d - 32] = x_mark[((size_t)b * L_SEQ + t) * N_MARK + (d - 32)];
  __syncthreads();
  float val = 0.f;
#pragma unroll
  for (int k = 0; k < 3; ++k)
#pragma unroll
    for (int c = 0; c < C_IN; ++c)
      val = fmaf(s_xn[k * C_IN + c], token_w[((size_t)d * C_IN + c) * 3 + k], val);
#pragma unroll
  for (int j = 0; j < N_MARK; ++j) val = fmaf(s_mk[j], temp_w[d * N_MARK + j], val);
  int i = d >> 1;
  float f = __expf(-(float)(2 * i) * (9.210340371976184f / (float)D_MODEL)); // ln(1e4)/512
  float ang = (float)t * f;
  val += (d & 1) ? __cosf(ang) : __sinf(ang);
  xemb_bf[(size_t)bt * D_MODEL + d] = f2bf(val);
}

// ---------------- WMMA bf16 GEMM: C[M,N] = A[M,K] * W[N,K]^T ----------------
// One wave per 32x64 tile: 2 A fragments + 4 B fragments -> 8 WMMAs per K-step
// (8 * 16*16*32*2 FLOP per 12 b128 loads ~ 21 FLOP / L2-byte).
// Layouts: A lane row = lane&15, 16 contiguous K at (lane>=16)*16;
//          B lane col = lane&15, 16 contiguous K at (lane>=16)*16.
// EPI==1: out = softplus(out + bias[col]).  REMAP: A row m -> (m/rpb)*bstride + toff + m%rpb.
template <int EPI, bool REMAP>
__global__ void __launch_bounds__(32)
gemm_bf16_wmma(const u16t* __restrict__ A, const u16t* __restrict__ W,
               float* __restrict__ C, const float* __restrict__ bias,
               int M, int N, int K, int n_off,
               int rows_per_batch, int batch_stride, int t_off) {
  const int lane   = threadIdx.x & 31;
  const int laneLo = lane & 15;
  const int hi     = lane >> 4;
  const int m0 = blockIdx.x * 32;
  const int n0 = blockIdx.y * 64;

  int am0 = m0 + laneLo;                      // A fragment rows for this lane
  int am1 = m0 + 16 + laneLo;
  if constexpr (REMAP) {
    am0 = (am0 / rows_per_batch) * batch_stride + t_off + (am0 % rows_per_batch);
    am1 = (am1 / rows_per_batch) * batch_stride + t_off + (am1 % rows_per_batch);
  }
  const u16t* __restrict__ Arow0 = A + (size_t)am0 * K;
  const u16t* __restrict__ Arow1 = A + (size_t)am1 * K;

  v8f acc[8] = {};
  const int kbA = hi * 16;
  const int kbB = hi * 16;
  for (int kk = 0; kk < K; kk += 32) {
    v16bf a0 = load_a_frag(Arow0 + kk + kbA);
    v16bf a1 = load_a_frag(Arow1 + kk + kbA);
#pragma unroll
    for (int s = 0; s < 4; ++s) {
      const u16t* Wrow = W + (size_t)(n_off + n0 + 16 * s + laneLo) * K + kk + kbB;
      v16bf bfr = load_b_frag(Wrow);
      acc[s] = __builtin_amdgcn_wmma_f32_16x16x32_bf16(
          false, a0, false, bfr, (short)0, acc[s], false, false);
      acc[4 + s] = __builtin_amdgcn_wmma_f32_16x16x32_bf16(
          false, a1, false, bfr, (short)0, acc[4 + s], false, false);
    }
  }
#pragma unroll
  for (int s = 0; s < 4; ++s) {
    int ocol = n0 + 16 * s + laneLo;
#pragma unroll
    for (int r = 0; r < 8; ++r) {
      float v0 = acc[s][r];
      float v1 = acc[4 + s][r];
      if constexpr (EPI == 1) {
        float bsv = bias[ocol];
        v0 = softplusf_(v0 + bsv);
        v1 = softplusf_(v1 + bsv);
      }
      C[(size_t)(m0 + r + 8 * hi) * N + ocol]      = v0;
      C[(size_t)(m0 + 16 + r + 8 * hi) * N + ocol] = v1;
    }
  }
}

// ---------------- causal depthwise conv(k=4) + SiLU, f32 + bf16 out ----------------
__global__ void conv_silu_kernel(const float* __restrict__ u_pre, const float* __restrict__ conv_w,
                                 const float* __restrict__ conv_b, float* __restrict__ usil,
                                 u16t* __restrict__ usil_bf) {
  int bt = blockIdx.x; int b = bt / L_SEQ, t = bt % L_SEQ;
  for (int j = 0; j < D_INNER / 256; ++j) {
    int d = threadIdx.x + j * 256;
    float acc = conv_b[d];
#pragma unroll
    for (int k = 0; k < D_CONV; ++k) {
      int tt = t + k - (D_CONV - 1);
      if (tt >= 0)
        acc = fmaf(conv_w[d * D_CONV + k], u_pre[((size_t)b * L_SEQ + tt) * D_INNER + d], acc);
    }
    float v = siluf_(acc);
    size_t o = (size_t)bt * D_INNER + d;
    usil[o] = v; usil_bf[o] = f2bf(v);
  }
}

// ---------------- selective scan: h = exp(dt*A)h + dt*B*u ; y = <h,C> ----------------
// One wave per 32 channels of one batch, barrier-free: lane n loads Bm/Cm element n
// (32 floats = exactly one wave32 b32 load) and the unrolled state loop broadcasts them
// with v_readlane (SGPR broadcast) instead of LDS + block barriers. Prefetch (distance 8)
// hides the load->exp->fma chain of this inherently sequential loop.
__global__ void __launch_bounds__(32)
mamba_scan(const float* __restrict__ dtv, const float* __restrict__ usil,
           const float* __restrict__ dbl, const float* __restrict__ z96,
           const float* __restrict__ A_log, const float* __restrict__ Dskip,
           u16t* __restrict__ y_bf) {
  int b = blockIdx.y;
  int lane = threadIdx.x & 31;
  int d = blockIdx.x * 32 + lane;
  float a[D_STATE], h[D_STATE];
#pragma unroll
  for (int n = 0; n < D_STATE; ++n) {
    a[n] = -__expf(A_log[(size_t)d * D_STATE + n]);
    h[n] = 0.f;
  }
  float dsk = Dskip[d];
  for (int t = 0; t < L_SEQ; ++t) {
    float bc = dbl[((size_t)b * L_SEQ + t) * 64 + 32 + lane];   // lanes 0..15: Bm, 16..31: Cm
    size_t idx = ((size_t)b * L_SEQ + t) * D_INNER + d;
    float dt = dtv[idx];
    float uu = usil[idx];
    if (t + 8 < L_SEQ) {
      __builtin_prefetch(&dbl[((size_t)b * L_SEQ + t + 8) * 64 + 32 + lane], 0, 0);
      __builtin_prefetch(&dtv[idx + (size_t)8 * D_INNER], 0, 0);
      __builtin_prefetch(&usil[idx + (size_t)8 * D_INNER], 0, 0);
    }
    float y = 0.f;
    u32t bcu = __float_as_uint(bc);
#pragma unroll
    for (int n = 0; n < D_STATE; ++n) {
      float bn = __uint_as_float(__builtin_amdgcn_readlane(bcu, n));
      float cn = __uint_as_float(__builtin_amdgcn_readlane(bcu, 16 + n));
      float dA = __expf(dt * a[n]);              // dt>0, a<0 -> (0,1]
      h[n] = fmaf(dA, h[n], dt * bn * uu);
      y = fmaf(h[n], cn, y);
    }
    if (t >= T_OFF) {
      int w = t - T_OFF;
      size_t orow = (size_t)(b * PRED_LEN + w);
      float z = z96[orow * D_INNER + d];
      y = (y + uu * dsk) * siluf_(z);
      y_bf[orow * D_INNER + d] = f2bf(y);
    }
  }
}

// ---------------- head projection (512->7) + de-normalization ----------------
__global__ void head_denorm(const float* __restrict__ o1, const float* __restrict__ head_w,
                            const float* __restrict__ meanv, const float* __restrict__ stdv,
                            float* __restrict__ out) {
  int idx = blockIdx.x * blockDim.x + threadIdx.x;
  if (idx >= M_TAIL * C_OUT) return;
  int row = idx / C_OUT, c = idx % C_OUT;
  int b = row / PRED_LEN;
  const float* orow = o1 + (size_t)row * D_MODEL;
  const float* wrow = head_w + (size_t)c * D_MODEL;
  float acc = 0.f;
  for (int d = 0; d < D_MODEL; ++d) acc = fmaf(orow[d], wrow[d], acc);
  out[idx] = acc * stdv[b * C_IN + c] + meanv[b * C_IN + c];
}

// ---------------- host orchestration ----------------
extern "C" void kernel_launch(void* const* d_in, const int* in_sizes, int n_in,
                              void* d_out, int out_size, void* d_ws, size_t ws_size,
                              hipStream_t stream) {
  (void)in_sizes; (void)n_in; (void)out_size; (void)ws_size;
  const float* x_enc      = (const float*)d_in[0];
  const float* x_mark_enc = (const float*)d_in[1];
  const float* token_w    = (const float*)d_in[4];
  const float* temp_w     = (const float*)d_in[5];
  const float* in_proj_w  = (const float*)d_in[6];
  const float* conv_w     = (const float*)d_in[7];
  const float* conv_b     = (const float*)d_in[8];
  const float* x_proj_w   = (const float*)d_in[9];
  const float* dt_w       = (const float*)d_in[10];
  const float* dt_b       = (const float*)d_in[11];
  const float* A_log      = (const float*)d_in[12];
  const float* Dskip      = (const float*)d_in[13];
  const float* out_proj_w = (const float*)d_in[14];
  const float* head_w     = (const float*)d_in[15];
  float* out = (float*)d_out;

  char* ws = (char*)d_ws;
  size_t off = 0;
  auto alloc = [&](size_t bytes) -> void* {
    void* p = ws + off; off += (bytes + 255) & ~(size_t)255; return p;
  };
  float* meanv      = (float*)alloc(B_SZ * C_IN * 4);
  float* stdv       = (float*)alloc(B_SZ * C_IN * 4);
  u16t* xemb_bf     = (u16t*)alloc((size_t)M_ROWS * D_MODEL * 2);          // 8 MB
  u16t* w_inproj_bf = (u16t*)alloc((size_t)2 * D_INNER * D_MODEL * 2);     // 2 MB
  u16t* w_xproj_bf  = (u16t*)alloc((size_t)(DT_RANK + 2 * D_STATE) * D_INNER * 2);
  u16t* w_dt_bf     = (u16t*)alloc((size_t)D_INNER * DT_RANK * 2);
  u16t* w_outproj_bf= (u16t*)alloc((size_t)D_MODEL * D_INNER * 2);
  float* u_pre      = (float*)alloc((size_t)M_ROWS * D_INNER * 4);         // 32 MB
  float* z96        = (float*)alloc((size_t)M_TAIL * D_INNER * 4);
  float* usil       = (float*)alloc((size_t)M_ROWS * D_INNER * 4);         // 32 MB
  u16t* usil_bf     = (u16t*)alloc((size_t)M_ROWS * D_INNER * 2);          // 16 MB
  float* dbl        = (float*)alloc((size_t)M_ROWS * 64 * 4);              // 2 MB
  u16t* dt_bf       = (u16t*)alloc((size_t)M_ROWS * DT_RANK * 2);
  float* dtv        = (float*)alloc((size_t)M_ROWS * D_INNER * 4);         // 32 MB
  u16t* y_bf        = (u16t*)alloc((size_t)M_TAIL * D_INNER * 2);
  float* o1         = (float*)alloc((size_t)M_TAIL * D_MODEL * 4);

  // 1) instance-norm statistics
  instnorm_stats<<<B_SZ * C_IN, 256, 0, stream>>>(x_enc, meanv, stdv);

  // 2) weight conversions to bf16
  {
    int n;
    n = 2 * D_INNER * D_MODEL;
    f32_to_bf16<<<(n + 255) / 256, 256, 0, stream>>>(in_proj_w, w_inproj_bf, n);
    n = (DT_RANK + 2 * D_STATE) * D_INNER;
    f32_to_bf16<<<(n + 255) / 256, 256, 0, stream>>>(x_proj_w, w_xproj_bf, n);
    n = D_INNER * DT_RANK;
    f32_to_bf16<<<(n + 255) / 256, 256, 0, stream>>>(dt_w, w_dt_bf, n);
    n = D_MODEL * D_INNER;
    f32_to_bf16<<<(n + 255) / 256, 256, 0, stream>>>(out_proj_w, w_outproj_bf, n);
  }

  // 3) fused embedding -> bf16 activations
  embed_kernel<<<M_ROWS, D_MODEL, 0, stream>>>(x_enc, x_mark_enc, token_w, temp_w,
                                               meanv, stdv, xemb_bf);

  // 4) in_proj u-half: (8192,512) x (1024,512)^T -> u_pre
  gemm_bf16_wmma<0, false><<<dim3(M_ROWS / 32, D_INNER / 64), 32, 0, stream>>>(
      xemb_bf, w_inproj_bf, u_pre, nullptr, M_ROWS, D_INNER, D_MODEL, 0, 1, 0, 0);

  // 5) in_proj z-half only for the 96 tail steps per batch (row-remapped A)
  gemm_bf16_wmma<0, true><<<dim3(M_TAIL / 32, D_INNER / 64), 32, 0, stream>>>(
      xemb_bf, w_inproj_bf, z96, nullptr, M_TAIL, D_INNER, D_MODEL, D_INNER,
      PRED_LEN, L_SEQ, T_OFF);

  // 6) causal depthwise conv + SiLU
  conv_silu_kernel<<<M_ROWS, 256, 0, stream>>>(u_pre, conv_w, conv_b, usil, usil_bf);

  // 7) x_proj: (8192,1024) x (64,1024)^T -> dbl (dt_low | B | C)
  gemm_bf16_wmma<0, false><<<dim3(M_ROWS / 32, 1), 32, 0, stream>>>(
      usil_bf, w_xproj_bf, dbl, nullptr, M_ROWS, 64, D_INNER, 0, 1, 0, 0);

  // 8) dt_low slice -> bf16
  {
    int n = M_ROWS * DT_RANK;
    slice32of64_to_bf16<<<(n + 255) / 256, 256, 0, stream>>>(dbl, dt_bf, n);
  }

  // 9) dt GEMM (K=32, single WMMA step) + fused bias + softplus
  gemm_bf16_wmma<1, false><<<dim3(M_ROWS / 32, D_INNER / 64), 32, 0, stream>>>(
      dt_bf, w_dt_bf, dtv, dt_b, M_ROWS, D_INNER, DT_RANK, 0, 1, 0, 0);

  // 10) selective scan (sequential in t), barrier-free, emits gated y tail as bf16
  mamba_scan<<<dim3(D_INNER / 32, B_SZ), 32, 0, stream>>>(
      dtv, usil, dbl, z96, A_log, Dskip, y_bf);

  // 11) out_proj on tail rows: (384,1024) x (512,1024)^T -> o1
  gemm_bf16_wmma<0, false><<<dim3(M_TAIL / 32, D_MODEL / 64), 32, 0, stream>>>(
      y_bf, w_outproj_bf, o1, nullptr, M_TAIL, D_MODEL, D_INNER, 0, 1, 0, 0);

  // 12) head (512->7) + de-normalization
  {
    int n = M_TAIL * C_OUT;
    head_denorm<<<(n + 255) / 256, 256, 0, stream>>>(o1, head_w, meanv, stdv, out);
  }
}
